// BMOJOAttention_16690242912565
// MI455X (gfx1250) — matrix operand order repaired and checked by reading, not compiled, not run on hardware
//
#include <hip/hip_runtime.h>
#include <math.h>

typedef __bf16 bf16;
typedef __attribute__((ext_vector_type(8)))  __bf16 v8bf;
typedef __attribute__((ext_vector_type(16))) __bf16 v16bf;
typedef __attribute__((ext_vector_type(8)))  float  v8f;
typedef __attribute__((ext_vector_type(4)))  unsigned int v4u;
typedef __attribute__((ext_vector_type(8)))  int v8i;
typedef __attribute__((ext_vector_type(4)))  int v4i;

union VV { v16bf v; v8bf h[2]; };

#define B_    2
#define S_    1024
#define DM_   2048
#define H_    16
#define HKV_  8
#define D_    128
#define G_    2
#define W_    256
#define EPS_  1e-6f
#define NEG_  (-1e30f)
#define SCALE_ 0.08838834764831845f   // 128^-0.5

// Tensor Data Mover path: probe-confirmed builtin; arity differs per toolchain.
#if defined(__HIP_DEVICE_COMPILE__)
#if __has_builtin(__builtin_amdgcn_tensor_load_to_lds) && __has_builtin(__builtin_amdgcn_s_wait_tensorcnt)
#define USE_TDM 1
#endif
#endif

__device__ __forceinline__ v8f zero8() {
    v8f z;
#pragma unroll
    for (int i = 0; i < 8; i++) z[i] = 0.f;
    return z;
}

#ifdef USE_TDM
// Issue one 2D TDM tile load: rows x cols bf16 elements from row-major global
// (leading dim ld_elems) into LDS at ldst. 16B of LDS padding (pad_amount code 3 =
// 4 DWORDs) is inserted after every (2 << pad_code) DWORDs:
//   pad_code 4 -> every 128B (64-elem rows -> 144B stride)
//   pad_code 5 -> every 256B (128-elem rows -> 272B stride)
__device__ __forceinline__ void tdm_load_tile(const bf16* gsrc, const bf16* ldst,
                                              int rows, int cols, int ld_elems,
                                              int pad_code) {
    unsigned long long ga = (unsigned long long)(size_t)gsrc;   // 57-bit global byte addr
    unsigned lo = (unsigned)(size_t)ldst;                       // LDS byte offset
    v4u g0;
    g0[0] = 1u;                                                 // count=1, user mode
    g0[1] = lo;                                                 // lds_addr
    g0[2] = (unsigned)ga;                                       // global_addr[31:0]
    g0[3] = ((unsigned)(ga >> 32) & 0x1FFFFFFu) | (2u << 30);   // global_addr[56:32] | type=2
    v8i g1;
    g1[0] = (int)((1u << 16)                    // data_size = 2 bytes
                | (1u << 20)                    // pad_enable
                | (((unsigned)pad_code) << 22)  // pad_interval
                | (3u << 25));                  // pad_amount: 4 DWORDs (=16B)
    g1[1] = (int)(((unsigned)ld_elems & 0xFFFFu) << 16);        // tensor_dim0[15:0]
    g1[2] = (int)((((unsigned)ld_elems >> 16) & 0xFFFFu)        // tensor_dim0[31:16]
                | (((unsigned)rows & 0xFFFFu) << 16));          // tensor_dim1[15:0]
    g1[3] = (int)(((unsigned)cols & 0xFFFFu) << 16);            // tile_dim0
    g1[4] = (int)((unsigned)rows & 0xFFFFu);                    // tile_dim1 (tile_dim2=0)
    g1[5] = ld_elems;                                           // tensor_dim0_stride[31:0]
    g1[6] = 0;
    g1[7] = 0;
    v4i gz; gz[0] = gz[1] = gz[2] = gz[3] = 0;
#if __clang_major__ >= 23
    v8i gz8; for (int i = 0; i < 8; i++) gz8[i] = 0;
    __builtin_amdgcn_tensor_load_to_lds(g0, g1, gz, gz, gz8, 0);
#else
    __builtin_amdgcn_tensor_load_to_lds(g0, g1, gz, gz, 0);
#endif
}
#endif

// ---------------------------------------------------------------- f32 -> bf16
__global__ __launch_bounds__(256) void k_cvt_bf16(const float* __restrict__ in,
                                                  bf16* __restrict__ out, int n) {
    int i = blockIdx.x * 256 + threadIdx.x;
    if (i < n) out[i] = (bf16)in[i];
}

// ------------------------------------------------------- WMMA GEMM  C = A*W^T + bias
// A: (M,K) bf16 row-major. W: (N,K) bf16 row-major. C: (M,N) f32.
// Block tile 128x128, K-tile 64, 8 waves (2x4), each wave 64x32 via 4x2 wmma accums.
// K-loop staging: TDM double-buffered tile DMA (wave 0 drives the Tensor Data Mover),
// falling back to cooperative global->LDS staging if the builtin is unavailable.
#define BM 128
#define BN 128
#define BK 64
__global__ __launch_bounds__(256)
void k_gemm_bf16(const bf16* __restrict__ A, const bf16* __restrict__ Wt,
                 const float* __restrict__ bias, float* __restrict__ C,
                 int M, int N, int K) {
    __shared__ alignas(16) bf16 As[2][BM][BK + 8];
    __shared__ alignas(16) bf16 Bs[2][BM][BK + 8];
    const int tid = threadIdx.x;
    const int lane = tid & 31;
    const int w    = tid >> 5;   // 0..7
    const int wm   = w >> 2;     // 0..1 -> 64 rows
    const int wn   = w & 3;      // 0..3 -> 32 cols
    const int m0 = blockIdx.y * BM;
    const int n0 = blockIdx.x * BN;
    const int hi  = (lane >= 16) ? 1 : 0;
    const int l16 = lane & 15;

    v8f acc[4][2];
#pragma unroll
    for (int mi = 0; mi < 4; mi++)
#pragma unroll
        for (int ni = 0; ni < 2; ni++) acc[mi][ni] = zero8();

#ifdef USE_TDM
    if (w == 0) {
        tdm_load_tile(A  + (size_t)m0 * K, &As[0][0][0], BM, BK, K, 4);
        tdm_load_tile(Wt + (size_t)n0 * K, &Bs[0][0][0], BM, BK, K, 4);
    }
    int buf = 0;
    for (int k0 = 0; k0 < K; k0 += BK, buf ^= 1) {
        if (w == 0) __builtin_amdgcn_s_wait_tensorcnt(0);
        __syncthreads();
        if (w == 0 && (k0 + BK) < K) {
            tdm_load_tile(A  + (size_t)m0 * K + k0 + BK, &As[buf ^ 1][0][0], BM, BK, K, 4);
            tdm_load_tile(Wt + (size_t)n0 * K + k0 + BK, &Bs[buf ^ 1][0][0], BM, BK, K, 4);
        }
#else
    int buf = 0;
    for (int k0 = 0; k0 < K; k0 += BK) {
#pragma unroll
        for (int c = tid; c < (BM * BK / 8); c += 256) {
            int row = c >> 3;
            int col = (c & 7) * 8;
            *(v8bf*)(&As[0][row][col]) = *(const v8bf*)(A  + (size_t)(m0 + row) * K + k0 + col);
            *(v8bf*)(&Bs[0][row][col]) = *(const v8bf*)(Wt + (size_t)(n0 + row) * K + k0 + col);
        }
        __syncthreads();
#endif
#pragma unroll
        for (int kk = 0; kk < BK; kk += 32) {
            v16bf af[4], bfv[2];
            const int abase = kk + (hi ? 8 : 0);
#pragma unroll
            for (int mi = 0; mi < 4; mi++) {
                int row = wm * 64 + mi * 16 + l16;
                VV u;
                u.h[0] = *(const v8bf*)(&As[buf][row][abase]);
                u.h[1] = *(const v8bf*)(&As[buf][row][abase + 16]);
                af[mi] = u.v;
            }
            const int bbase = kk + (hi ? 8 : 0);
#pragma unroll
            for (int ni = 0; ni < 2; ni++) {
                int row = wn * 32 + ni * 16 + l16;
                VV u;
                u.h[0] = *(const v8bf*)(&Bs[buf][row][bbase]);
                u.h[1] = *(const v8bf*)(&Bs[buf][row][bbase + 16]);
                bfv[ni] = u.v;
            }
#pragma unroll
            for (int mi = 0; mi < 4; mi++)
#pragma unroll
                for (int ni = 0; ni < 2; ni++)
                    acc[mi][ni] = __builtin_amdgcn_wmma_f32_16x16x32_bf16(
                        false, af[mi], false, bfv[ni], (short)0, acc[mi][ni], false, false);
        }
#ifndef USE_TDM
        __syncthreads();
#endif
    }
#pragma unroll
    for (int mi = 0; mi < 4; mi++)
#pragma unroll
        for (int ni = 0; ni < 2; ni++) {
            int col = n0 + wn * 32 + ni * 16 + l16;
            float bv = bias ? bias[col] : 0.f;
#pragma unroll
            for (int r = 0; r < 8; r++) {
                int row = m0 + wm * 64 + mi * 16 + r + (hi ? 8 : 0);
                C[(size_t)row * N + col] = acc[mi][ni][r] + bv;
            }
        }
}

// ------------------------------------------ rmsnorm + rope, f32(B,S,NH,D) -> bf16(B,NH,S,D)
__global__ __launch_bounds__(128)
void k_qk_post(const float* __restrict__ raw, const float* __restrict__ g,
               const float* __restrict__ cosp, const float* __restrict__ sinp,
               bf16* __restrict__ out, int NH) {
    int rowid = blockIdx.x;              // b*S*NH + s*NH + h
    int h = rowid % NH;
    int s = (rowid / NH) % S_;
    int b = rowid / (NH * S_);
    int d = threadIdx.x;                 // 0..127
    __shared__ float xs[128];
    __shared__ float red[128];
    float x = raw[(size_t)rowid * D_ + d];
    red[d] = x * x;
    __syncthreads();
    for (int off = 64; off > 0; off >>= 1) {
        if (d < off) red[d] += red[d + off];
        __syncthreads();
    }
    float inv = rsqrtf(red[0] * (1.f / D_) + EPS_);
    float xn = x * inv * g[d];
    xs[d] = xn;
    __syncthreads();
    float rot = (d < 64) ? -xs[d + 64] : xs[d - 64];
    size_t cid = ((size_t)b * S_ + s) * D_ + d;
    float y = xn * cosp[cid] + rot * sinp[cid];
    out[(((size_t)b * NH + h) * S_ + s) * D_ + d] = (bf16)y;
}

// --------------------------------------- V: f32(B,S,HKV,D) -> bf16 transposed (B,HKV,D,S)
__global__ __launch_bounds__(256)
void k_v_post(const float* __restrict__ raw, bf16* __restrict__ outT) {
    size_t idx = (size_t)blockIdx.x * 256 + threadIdx.x;
    if (idx >= (size_t)B_ * S_ * HKV_ * D_) return;
    int d = idx % D_;
    int h = (idx / D_) % HKV_;
    int s = (idx / ((size_t)D_ * HKV_)) % S_;
    int b = idx / ((size_t)D_ * HKV_ * S_);
    outT[(((size_t)b * HKV_ + h) * D_ + d) * S_ + s] = (bf16)raw[idx];
}

// ------------------------------------------------- attention helpers (LDS-staged tiles)
#define KS_LDW 136   // 128-elem K rows + 8-elem pad  (272B stride, conflict-free B reads)
#define VS_LDW 72    // 64-elem V rows + 8-elem pad   (144B stride)

// Scores: S[16q x 64k]; K tile staged in LDS as [64][KS_LDW] (row = key-ks, col = d)
__device__ __forceinline__ void qk_scores(const v16bf aq[4], const bf16* __restrict__ Ks,
                                          int hi, int l16, v8f s[4]) {
#pragma unroll
    for (int n = 0; n < 4; n++) {
        s[n] = zero8();
        int row = n * 16 + l16;          // key (tile-relative) for this lane
#pragma unroll
        for (int kc = 0; kc < 4; kc++) { // D in chunks of 32
            int base = kc * 32 + (hi ? 8 : 0);
            VV u;
            u.h[0] = *(const v8bf*)(Ks + (size_t)row * KS_LDW + base);
            u.h[1] = *(const v8bf*)(Ks + (size_t)row * KS_LDW + base + 16);
            s[n] = __builtin_amdgcn_wmma_f32_16x16x32_bf16(
                false, aq[kc], false, u.v, (short)0, s[n], false, false);
        }
    }
}

// O += P(16x64) * V(64x128); P per-wave LDS [16][72]; V tile in LDS [128][VS_LDW]
__device__ __forceinline__ void pv_accum(const bf16* __restrict__ Ps, const bf16* __restrict__ Vs,
                                         int hi, int l16, v8f acc[8]) {
    v16bf pa[2];
    const int abase = hi ? 8 : 0;
#pragma unroll
    for (int c = 0; c < 2; c++) {
        VV u;
        u.h[0] = *(const v8bf*)(Ps + (size_t)l16 * 72 + c * 32 + abase);
        u.h[1] = *(const v8bf*)(Ps + (size_t)l16 * 72 + c * 32 + abase + 16);
        pa[c] = u.v;
    }
#pragma unroll
    for (int dt = 0; dt < 8; dt++) {
        int dcol = dt * 16 + l16;        // B operand: col = output d
#pragma unroll
        for (int c = 0; c < 2; c++) {
            int jrel = c * 32 + (hi ? 8 : 0);
            VV u;
            u.h[0] = *(const v8bf*)(Vs + (size_t)dcol * VS_LDW + jrel);
            u.h[1] = *(const v8bf*)(Vs + (size_t)dcol * VS_LDW + jrel + 16);
            acc[dt] = __builtin_amdgcn_wmma_f32_16x16x32_bf16(
                false, pa[c], false, u.v, (short)0, acc[dt], false, false);
        }
    }
}

// ----------------------------------------------- dual-branch flash attention
// grid (S/64, H, B), block 128 (4 waves); wave w owns query rows [qs+16w, qs+16w+16)
// K/V tiles for the needed branches are DMA'd into LDS by the TDM (wave 0), shared by
// all 4 waves; P-transpose stays in per-wave LDS regions.
__global__ __launch_bounds__(128)
void k_attn(const bf16* __restrict__ qic, const bf16* __restrict__ kic, const bf16* __restrict__ vic,
            const bf16* __restrict__ qf,  const bf16* __restrict__ kf,  const bf16* __restrict__ vf,
            bf16* __restrict__ obf) {
    const int qt = blockIdx.x, h = blockIdx.y, b = blockIdx.z;
    const int qs = qt * 64;
    const int kvh = h / G_;
    const int tid = threadIdx.x;
    const int lane = tid & 31;
    const int w = tid >> 5;
    const int hi = (lane >= 16) ? 1 : 0;
    const int l16 = lane & 15;

    const bf16* Qi = qic + ((size_t)b * H_ + h) * S_ * D_;
    const bf16* Qf = qf  + ((size_t)b * H_ + h) * S_ * D_;
    const bf16* Ki = kic + ((size_t)b * HKV_ + kvh) * S_ * D_;
    const bf16* Kf = kf  + ((size_t)b * HKV_ + kvh) * S_ * D_;
    const bf16* Vi = vic + ((size_t)b * HKV_ + kvh) * D_ * S_;   // (D,S)
    const bf16* Vf = vf  + ((size_t)b * HKV_ + kvh) * D_ * S_;

    __shared__ alignas(16) bf16 Kis[64][KS_LDW];
    __shared__ alignas(16) bf16 Kfs[64][KS_LDW];
    __shared__ alignas(16) bf16 Vis[128][VS_LDW];
    __shared__ alignas(16) bf16 Vfs[128][VS_LDW];
    __shared__ alignas(16) bf16 Pi_s[4][16][72];
    __shared__ alignas(16) bf16 Pf_s[4][16][72];

    // Q fragments for both branches (rows qs+16w..+15, 4 chunks of K=32 over D)
    v16bf aqi[4], aqf[4];
    {
        int row = qs + w * 16 + l16;
        int base = hi ? 8 : 0;
#pragma unroll
        for (int kc = 0; kc < 4; kc++) {
            VV u;
            u.h[0] = *(const v8bf*)(Qi + (size_t)row * D_ + kc * 32 + base);
            u.h[1] = *(const v8bf*)(Qi + (size_t)row * D_ + kc * 32 + base + 16);
            aqi[kc] = u.v;
            u.h[0] = *(const v8bf*)(Qf + (size_t)row * D_ + kc * 32 + base);
            u.h[1] = *(const v8bf*)(Qf + (size_t)row * D_ + kc * 32 + base + 16);
            aqf[kc] = u.v;
        }
    }

    v8f acc[8];
#pragma unroll
    for (int dt = 0; dt < 8; dt++) acc[dt] = zero8();
    float m[8], l[8];
#pragma unroll
    for (int r = 0; r < 8; r++) { m[r] = NEG_; l[r] = 0.f; }

    for (int kt = 0; kt <= qt; kt++) {
        const int ks = kt * 64;
        const int dq = qs - ks;
        // tile class: 0=diagonal(window+causal mask), 1=pure window, 2=mixed, 3=pure fading
        const int mode = (dq == 0) ? 0 : (dq <= 192) ? 1 : (dq == 256) ? 2 : 3;
        const bool doI = (mode <= 2), doF = (mode >= 2);

        // stage K/V tiles for the active branches
        __syncthreads();   // prior iteration's LDS reads must finish before overwrite
#ifdef USE_TDM
        if (w == 0) {
            if (doI) {
                tdm_load_tile(Ki + (size_t)ks * D_, &Kis[0][0], 64, 128, D_, 5);
                tdm_load_tile(Vi + ks,              &Vis[0][0], 128, 64, S_, 4);
            }
            if (doF) {
                tdm_load_tile(Kf + (size_t)ks * D_, &Kfs[0][0], 64, 128, D_, 5);
                tdm_load_tile(Vf + ks,              &Vfs[0][0], 128, 64, S_, 4);
            }
            __builtin_amdgcn_s_wait_tensorcnt(0);
        }
#else
        // cooperative staging fallback (128 threads)
        for (int c = tid; c < 64 * 128 / 8; c += 128) {
            int row = c >> 4, col = (c & 15) * 8;
            if (doI) *(v8bf*)(&Kis[row][col]) = *(const v8bf*)(Ki + (size_t)(ks + row) * D_ + col);
            if (doF) *(v8bf*)(&Kfs[row][col]) = *(const v8bf*)(Kf + (size_t)(ks + row) * D_ + col);
        }
        for (int c = tid; c < 128 * 64 / 8; c += 128) {
            int row = c >> 3, col = (c & 7) * 8;
            if (doI) *(v8bf*)(&Vis[row][col]) = *(const v8bf*)(Vi + (size_t)row * S_ + ks + col);
            if (doF) *(v8bf*)(&Vfs[row][col]) = *(const v8bf*)(Vf + (size_t)row * S_ + ks + col);
        }
#endif
        __syncthreads();

        v8f si[4], sf_[4];
#pragma unroll
        for (int n = 0; n < 4; n++) { si[n] = zero8(); sf_[n] = zero8(); }
        if (doI) qk_scores(aqi, &Kis[0][0], hi, l16, si);
        if (doF) qk_scores(aqf, &Kfs[0][0], hi, l16, sf_);

        // online softmax per owned row r  (lane row = r + 8*hi, col = ks + n*16 + l16)
#pragma unroll
        for (int r = 0; r < 8; r++) {
            const int i_row = qs + w * 16 + r + (hi ? 8 : 0);
            float sel[4];
#pragma unroll
            for (int n = 0; n < 4; n++) {
                int j = ks + n * 16 + l16;
                int dd = i_row - j;
                float sv;
                if      (mode == 0) sv = (dd >= 0) ? si[n][r] * SCALE_ : NEG_;
                else if (mode == 1) sv = si[n][r] * SCALE_;
                else if (mode == 2) sv = (dd >= W_) ? sf_[n][r] * SCALE_ : si[n][r] * SCALE_;
                else                sv = sf_[n][r] * SCALE_;
                sel[n] = sv;
            }
            float tmax = fmaxf(fmaxf(sel[0], sel[1]), fmaxf(sel[2], sel[3]));
#pragma unroll
            for (int off = 8; off >= 1; off >>= 1) tmax = fmaxf(tmax, __shfl_xor(tmax, off, 16));
            float mn = fmaxf(m[r], tmax);
            float alpha = __expf(m[r] - mn);
            float rsum = 0.f;
#pragma unroll
            for (int n = 0; n < 4; n++) {
                int j = ks + n * 16 + l16;
                int dd = i_row - j;
                float p = __expf(sel[n] - mn);
                rsum += p;
                bf16 pb = (bf16)p;
                if (doI) Pi_s[w][r + (hi ? 8 : 0)][n * 16 + l16] =
                    (mode == 2 && dd >= W_) ? (bf16)0.f : pb;
                if (doF) Pf_s[w][r + (hi ? 8 : 0)][n * 16 + l16] =
                    (mode == 2 && dd < W_) ? (bf16)0.f : pb;
            }
#pragma unroll
            for (int off = 8; off >= 1; off >>= 1) rsum += __shfl_xor(rsum, off, 16);
            l[r] = l[r] * alpha + rsum;
            m[r] = mn;
#pragma unroll
            for (int dt = 0; dt < 8; dt++) acc[dt][r] *= alpha;
        }
        // P * V (P region is per-wave; V tiles are block-shared LDS)
        if (doI) pv_accum(&Pi_s[w][0][0], &Vis[0][0], hi, l16, acc);
        if (doF) pv_accum(&Pf_s[w][0][0], &Vfs[0][0], hi, l16, acc);
    }

    // write O in (B,S,H,D) so the Wo GEMM sees row-major (M=B*S, K=H*D)
#pragma unroll
    for (int dt = 0; dt < 8; dt++) {
        int dcol = dt * 16 + l16;
#pragma unroll
        for (int r = 0; r < 8; r++) {
            int i_row = qs + w * 16 + r + (hi ? 8 : 0);
            obf[(((size_t)b * S_ + i_row) * H_ + h) * D_ + dcol] = (bf16)(acc[dt][r] / l[r]);
        }
    }
}

// ----------------------------------------------------------------- launch
extern "C" void kernel_launch(void* const* d_in, const int* in_sizes, int n_in,
                              void* d_out, int out_size, void* d_ws, size_t ws_size,
                              hipStream_t stream) {
    (void)in_sizes; (void)n_in; (void)out_size; (void)ws_size;
    const float* x    = (const float*)d_in[0];
    const float* cosp = (const float*)d_in[1];
    const float* sinp = (const float*)d_in[2];
    const float* Wq   = (const float*)d_in[3];
    const float* bq   = (const float*)d_in[4];
    const float* Wk   = (const float*)d_in[5];
    const float* bk   = (const float*)d_in[6];
    const float* Wv   = (const float*)d_in[7];
    const float* bv   = (const float*)d_in[8];
    const float* Wq2  = (const float*)d_in[9];
    const float* bq2  = (const float*)d_in[10];
    const float* Wk2  = (const float*)d_in[11];
    const float* bk2  = (const float*)d_in[12];
    const float* Wv2  = (const float*)d_in[13];
    const float* bv2  = (const float*)d_in[14];
    const float* Wo   = (const float*)d_in[15];
    const float* gq   = (const float*)d_in[16];
    const float* gk   = (const float*)d_in[17];
    const float* gq2  = (const float*)d_in[18];
    const float* gk2  = (const float*)d_in[19];
    float* out = (float*)d_out;

    char* ws = (char*)d_ws;
    size_t off = 0;
    auto alloc = [&](size_t bytes) -> void* {
        void* p = ws + off;
        off = (off + bytes + 255) & ~(size_t)255;
        return p;
    };
    const size_t nx   = (size_t)B_ * S_ * DM_;
    const size_t nqw  = (size_t)H_ * D_ * DM_;
    const size_t nkw  = (size_t)HKV_ * D_ * DM_;
    const size_t nq   = (size_t)B_ * S_ * H_ * D_;
    const size_t nk   = (size_t)B_ * S_ * HKV_ * D_;

    bf16* xbf  = (bf16*)alloc(nx * 2);
    bf16* wqB  = (bf16*)alloc(nqw * 2);
    bf16* wkB  = (bf16*)alloc(nkw * 2);
    bf16* wvB  = (bf16*)alloc(nkw * 2);
    bf16* wq2B = (bf16*)alloc(nqw * 2);
    bf16* wk2B = (bf16*)alloc(nkw * 2);
    bf16* wv2B = (bf16*)alloc(nkw * 2);
    bf16* woB  = (bf16*)alloc(nqw * 2);
    float* kraw = (float*)alloc(nk * 4);
    float* vraw = (float*)alloc(nk * 4);
    bf16* qicB = (bf16*)alloc(nq * 2);
    bf16* kicB = (bf16*)alloc(nk * 2);
    bf16* vicB = (bf16*)alloc(nk * 2);
    bf16* qfB  = (bf16*)alloc(nq * 2);
    bf16* kfB  = (bf16*)alloc(nk * 2);
    bf16* vfB  = (bf16*)alloc(nk * 2);
    bf16* obf  = (bf16*)alloc(nq * 2);
    float* qraw = out;   // d_out doubles as Q-raw scratch (same size), overwritten at the end

    auto cvt = [&](const float* src, bf16* dst, size_t n) {
        k_cvt_bf16<<<(unsigned)((n + 255) / 256), 256, 0, stream>>>(src, dst, (int)n);
    };
    cvt(x, xbf, nx);
    cvt(Wq, wqB, nqw);   cvt(Wk, wkB, nkw);   cvt(Wv, wvB, nkw);
    cvt(Wq2, wq2B, nqw); cvt(Wk2, wk2B, nkw); cvt(Wv2, wv2B, nkw);
    cvt(Wo, woB, nqw);

    const int M = B_ * S_;   // 2048
    const int K = DM_;       // 2048
    const int NQ = H_ * D_;  // 2048
    const int NK = HKV_ * D_;// 1024

    const bf16*  wqs[2] = { wqB, wq2B };
    const bf16*  wks[2] = { wkB, wk2B };
    const bf16*  wvs[2] = { wvB, wv2B };
    const float* bqs[2] = { bq, bq2 };
    const float* bks[2] = { bk, bk2 };
    const float* bvs[2] = { bv, bv2 };
    const float* gqs[2] = { gq, gq2 };
    const float* gks[2] = { gk, gk2 };
    bf16* qbs[2] = { qicB, qfB };
    bf16* kbs[2] = { kicB, kfB };
    bf16* vbs[2] = { vicB, vfB };

    for (int br = 0; br < 2; br++) {
        k_gemm_bf16<<<dim3(NQ / BN, M / BM), 256, 0, stream>>>(xbf, wqs[br], bqs[br], qraw, M, NQ, K);
        k_gemm_bf16<<<dim3(NK / BN, M / BM), 256, 0, stream>>>(xbf, wks[br], bks[br], kraw, M, NK, K);
        k_gemm_bf16<<<dim3(NK / BN, M / BM), 256, 0, stream>>>(xbf, wvs[br], bvs[br], vraw, M, NK, K);
        k_qk_post<<<B_ * S_ * H_, 128, 0, stream>>>(qraw, gqs[br], cosp, sinp, qbs[br], H_);
        k_qk_post<<<B_ * S_ * HKV_, 128, 0, stream>>>(kraw, gks[br], cosp, sinp, kbs[br], HKV_);
        k_v_post<<<(unsigned)((nk + 255) / 256), 256, 0, stream>>>(vraw, vbs[br]);
    }

    k_attn<<<dim3(S_ / 64, H_, B_), 128, 0, stream>>>(qicB, kicB, vicB, qfB, kfB, vfB, obf);

    // final projection: out = O @ Wo^T  (overwrites the q-raw scratch)
    k_gemm_bf16<<<dim3(DM_ / BN, M / BM), 256, 0, stream>>>(obf, woB, nullptr, out, M, DM_, K);
}
